// GTransformerHomo_67997922230897
// MI455X (gfx1250) — matrix operand order repaired and checked by reading, not compiled, not run on hardware
//
#include <hip/hip_runtime.h>
#include <hip/hip_bf16.h>

typedef __attribute__((ext_vector_type(16))) _Float16     v16h;
typedef __attribute__((ext_vector_type(8)))  float        v8f;
typedef __attribute__((ext_vector_type(4)))  unsigned int u32x4;
typedef __attribute__((ext_vector_type(8)))  int          i32x8;
typedef __attribute__((ext_vector_type(4)))  int          i32x4;

#define NN 50000
#define EE 800000
#define DD 128
#define HH 8
#define DH 16
#define EPS_SM   1e-8f
#define LN_EPS   1e-5f
#define NEG_SLP  0.01f

#if defined(__gfx1250__)
#define USE_TDM 1
#else
#define USE_TDM 0
#endif

// ---------------------------------------------------------------------------
// Fragment-native LDS swizzles (CDNA5 wave32 WMMA layouts, CDNA5 ISA §7.12.2)
// A 16x32 f16 frag: lane = ((k%32 bit3)<<4)|m ; elem i = (k&7) | ((k%32>>4)<<3)
// B 32x16 f16 frag: lane = ((k%32 bit3)<<4)|(n&15) ; elem i = (k&7) | ((k%32>>4)<<3)
// Each lane's 16 halfs are stored contiguously -> one 32B LDS read per frag.
// ---------------------------------------------------------------------------
__device__ __forceinline__ int a_swz_idx(int m, int k) {
  const int kt = k >> 5, kr = k & 31;
  const int lane = (((kr >> 3) & 1) << 4) | m;
  const int i    = (kr & 7) | ((kr >> 4) << 3);
  return ((kt * 32 + lane) << 4) | i;
}

__device__ __forceinline__ int b_swz_idx(int k, int n, int NT) {
  const int kt = k >> 5, kr = k & 31;
  const int lane = (((kr >> 3) & 1) << 4) | (n & 15);
  const int i    = (kr & 7) | ((kr >> 4) << 3);
  const int nt   = n >> 4;
  return (((kt * NT + nt) * 32 + lane) << 4) | i;
}

// ---------------------------------------------------------------------------
// Generic zero-fill (accumulators o and z)
// ---------------------------------------------------------------------------
__global__ void zero_f32(float* __restrict__ p, int n) {
  int i = blockIdx.x * blockDim.x + threadIdx.x;
  if (i < n) p[i] = 0.0f;
}

// ---------------------------------------------------------------------------
// C[M x 128] = A[M x 128] @ W[128 x 128] + bias.
// Block = 8 waves. The 64KB f32 weight tile is DMA'd global->LDS by the
// Tensor Data Mover (wave 0 issues TENSOR_LOAD_TO_LDS, waits TENSORcnt),
// converted to fragment-swizzled f16, then each wave owns one 16-row tile:
// 4 register-cached A frags x 8 col tiles = 32 WMMAs per wave.
// LDS: Bs 32KB + union(Wraw f32 64KB | per-wave A tiles 32KB) = 96KB.
// ---------------------------------------------------------------------------
template <bool HALF_OUT>
__global__ __launch_bounds__(256) void gemm_bias_128(const float* __restrict__ A,
                                                     const float* __restrict__ W,
                                                     const float* __restrict__ bias,
                                                     void* __restrict__ out,
                                                     int num_tiles) {
  __shared__ v16h Bs[4 * 8 * 32];                 // 32 KB swizzled weights
  __shared__ __align__(32) float Wraw[DD * DD];   // 64 KB union: raw W, then A tiles
  _Float16* Bh = (_Float16*)Bs;
  const int tid  = threadIdx.x;
  const int lane = tid & 31;
  const int wave = tid >> 5;

#if USE_TDM
  // --- TDM: DMA the 128x128 f32 weight tensor into LDS (one wave issues) ---
  if (wave == 0) {
    const unsigned long long ga = (unsigned long long)(uintptr_t)W;
    const unsigned int lds_off  = (unsigned int)(uintptr_t)(void*)Wraw; // addr[31:0] = LDS byte offset
    u32x4 g0;                                   // D# group 0 (CDNA5 ISA)
    g0[0] = 1u;                                 // count=1, user descriptor, no gather
    g0[1] = lds_off;                            // lds_addr
    g0[2] = (unsigned int)(ga & 0xffffffffull); // global_addr[31:0]
    g0[3] = (unsigned int)((ga >> 32) & 0x01ffffffull) | 0x80000000u; // addr[56:32] | type=2
    i32x8 g1;                                   // D# group 1 (§8.4)
    g1[0] = 0x00020000;                         // data_size=2 (4 bytes), mask=0, no flags
    g1[1] = (int)(128u << 16);                  // tensor_dim0 = 128 (bits 79:48 lo16)
    g1[2] = (int)(128u << 16);                  // tensor_dim1 = 128 (bits 111:80 lo16)
    g1[3] = (int)(128u << 16);                  // tile_dim0   = 128 (bits 127:112)
    g1[4] = 128;                                // tile_dim1   = 128 (bits 143:128)
    g1[5] = 128;                                // tensor_dim0_stride = 128 (bits 207:160 lo32)
    g1[6] = 0;
    g1[7] = 0;
    i32x4 gz; gz[0] = 0; gz[1] = 0; gz[2] = 0; gz[3] = 0;  // groups 2/3: unused (2D)
#if (__clang_major__ <= 22)
    __builtin_amdgcn_tensor_load_to_lds(g0, g1, gz, gz, 0);
#else
    i32x8 gz8; for (int ii = 0; ii < 8; ++ii) gz8[ii] = 0;
    __builtin_amdgcn_tensor_load_to_lds(g0, g1, gz, gz, gz8, 0);
#endif
    __builtin_amdgcn_s_wait_tensorcnt(0);       // s_wait_tensorcnt 0 (per-wave)
  }
  __syncthreads();
  for (int i = tid; i < DD * DD; i += 256) {    // LDS f32 -> swizzled f16 LDS
    const int k = i >> 7, n = i & 127;
    Bh[b_swz_idx(k, n, 8)] = (_Float16)Wraw[i];
  }
#else
  for (int i = tid; i < DD * DD; i += 256) {
    const int k = i >> 7, n = i & 127;
    Bh[b_swz_idx(k, n, 8)] = (_Float16)W[i];
  }
#endif
  __syncthreads();                              // Wraw now dead -> reuse for A tiles

  const int tile = blockIdx.x * 8 + wave;
  if (tile >= num_tiles) return;
  const int row0 = tile * 16;

  // per-wave A staging (swizzled); same-wave LDS ops are in-order
  _Float16* Aw = (_Float16*)((v16h*)Wraw + wave * (4 * 32));
  for (int i = lane; i < 16 * DD; i += 32) {
    const int m = i >> 7, k = i & 127;
    Aw[a_swz_idx(m, k)] = (_Float16)A[(size_t)row0 * DD + i];
  }
  const v16h* Av = (const v16h*)Aw;

  v16h afrag[4];
#pragma unroll
  for (int kt = 0; kt < 4; ++kt) afrag[kt] = Av[kt * 32 + lane];

  const int coln  = lane & 15;
  const int mbase = (lane >> 4) << 3;
#pragma unroll
  for (int nt = 0; nt < 8; ++nt) {
    v8f acc = {};
#pragma unroll
    for (int kt = 0; kt < 4; ++kt) {
      const v16h b = Bs[(kt * 8 + nt) * 32 + lane];
      acc = __builtin_amdgcn_wmma_f32_16x16x32_f16(false, afrag[kt], false, b,
                                                   (short)0, acc, false, false);
    }
    const int col  = nt * 16 + coln;
    const float bc = bias[col];
#pragma unroll
    for (int rr = 0; rr < 8; ++rr) {
      const int m      = mbase + rr;
      const float val  = acc[rr] + bc;
      const size_t off = (size_t)(row0 + m) * DD + col;
      if (HALF_OUT) ((_Float16*)out)[off] = (_Float16)val;
      else          ((float*)out)[off]    = val;
    }
  }
}

// ---------------------------------------------------------------------------
// Per-(edge,head) score: s = q[dst,h,:] . (k[src,h,:] + e[edge,h,:]),
// escore = exp(s), z[dst,h] += escore (atomic segment sum).
// ---------------------------------------------------------------------------
__global__ __launch_bounds__(256) void edge_score(const float* __restrict__ q,
                                                  const float* __restrict__ k,
                                                  const _Float16* __restrict__ e,
                                                  const int* __restrict__ src,
                                                  const int* __restrict__ dst,
                                                  float* __restrict__ z,
                                                  float* __restrict__ escore) {
  const int idx = blockIdx.x * blockDim.x + threadIdx.x;
  if (idx >= EE * HH) return;
  const int eid = idx >> 3;
  const int h   = idx & 7;
  const int sn  = src[eid];
  const int dn  = dst[eid];
  const float*    qp = q + (size_t)dn * DD + h * DH;
  const float*    kp = k + (size_t)sn * DD + h * DH;
  const _Float16* ep = e + (size_t)eid * DD + h * DH;
  __builtin_prefetch(ep, 0, 0);   // global_prefetch_b8
  float acc = 0.0f;
#pragma unroll
  for (int j = 0; j < DH; ++j) acc += qp[j] * (kp[j] + (float)ep[j]);
  const float es = __expf(acc);
  escore[idx] = es;
  atomicAdd(&z[dn * HH + h], es);
}

// ---------------------------------------------------------------------------
// Normalize escore in place (this region of d_out IS norm_escore) and scatter
// norm * (v[src] + e) into o[dst] with f32 global atomics.
// ---------------------------------------------------------------------------
__global__ __launch_bounds__(256) void edge_aggregate(const float* __restrict__ v,
                                                      const _Float16* __restrict__ e,
                                                      const int* __restrict__ src,
                                                      const int* __restrict__ dst,
                                                      const float* __restrict__ z,
                                                      float* __restrict__ escore_io,
                                                      float* __restrict__ o) {
  const int idx = blockIdx.x * blockDim.x + threadIdx.x;
  if (idx >= EE * HH) return;
  const int eid = idx >> 3;
  const int h   = idx & 7;
  const int sn  = src[eid];
  const int dn  = dst[eid];
  const float*    vp = v + (size_t)sn * DD + h * DH;
  const _Float16* ep = e + (size_t)eid * DD + h * DH;
  __builtin_prefetch(vp, 0, 0);   // global_prefetch_b8
  const float norm = 0.25f * escore_io[idx] / (EPS_SM + z[dn * HH + h]); // 1/sqrt(16)
  escore_io[idx] = norm;
  float* op = o + (size_t)dn * DD + h * DH;
#pragma unroll
  for (int j = 0; j < DH; ++j) atomicAdd(&op[j], norm * (vp[j] + (float)ep[j]));
}

// ---------------------------------------------------------------------------
// Fused gate: cat=[o,r,o-r] (16x384) @ Wg(384x128)+bg -> sigmoid gate,
// hmix = o + b*(r-o), LayerNorm over 128, leaky-ReLU -> x_out.
// LDS: Wg swz f16 (96 KB) + cat swz f16 (12 KB) + hmix f32 (8 KB) = 116 KB.
// ---------------------------------------------------------------------------
__global__ __launch_bounds__(256) void gate_norm_out(const float* __restrict__ o,
                                                     const float* __restrict__ r,
                                                     const float* __restrict__ Wg,
                                                     const float* __restrict__ bg,
                                                     const float* __restrict__ ln_g,
                                                     const float* __restrict__ ln_b,
                                                     float* __restrict__ x_out) {
  __shared__ v16h  Wgs[12 * 8 * 32];   // 96 KB swizzled Wg (384x128 f16)
  __shared__ v16h  cats[12 * 32];      // 12 KB swizzled cat (16x384 f16)
  __shared__ float hmix[16 * DD];      //  8 KB
  _Float16* Wh = (_Float16*)Wgs;
  _Float16* ch = (_Float16*)cats;
  const int tid  = threadIdx.x;
  const int lane = tid & 31;
  const int wave = tid >> 5;
  const int row0 = blockIdx.x * 16;

  for (int i = tid; i < 3 * DD * DD; i += 256) {
    const int k = i >> 7, n = i & 127;
    Wh[b_swz_idx(k, n, 8)] = (_Float16)Wg[i];
  }
  for (int i = tid; i < 16 * DD; i += 256) {
    const int m = i >> 7, n = i & 127;
    const float ov = o[(size_t)(row0 + m) * DD + n];
    const float rv = r[(size_t)(row0 + m) * DD + n];
    ch[a_swz_idx(m, n)]          = (_Float16)ov;
    ch[a_swz_idx(m, n + DD)]     = (_Float16)rv;
    ch[a_swz_idx(m, n + 2 * DD)] = (_Float16)(ov - rv);
  }
  __syncthreads();

  const int nt = wave;
  v8f acc = {};
#pragma unroll
  for (int kt = 0; kt < 12; ++kt) {
    const v16h a = cats[kt * 32 + lane];
    const v16h b = Wgs[(kt * 8 + nt) * 32 + lane];
    acc = __builtin_amdgcn_wmma_f32_16x16x32_f16(false, a, false, b,
                                                 (short)0, acc, false, false);
  }

  const int col   = nt * 16 + (lane & 15);
  const int mbase = (lane >> 4) << 3;
  const float bgc = bg[col];
#pragma unroll
  for (int rr = 0; rr < 8; ++rr) {
    const int m = mbase + rr;
    const float g    = acc[rr] + bgc;
    const float bsig = 1.0f / (1.0f + __expf(-g));
    const float ov   = (float)ch[a_swz_idx(m, col)];
    const float rv   = (float)ch[a_swz_idx(m, col + DD)];
    hmix[m * DD + col] = ov + bsig * (rv - ov);
  }
  __syncthreads();

  // LayerNorm: 16 threads per row (rows 2w / 2w+1 live in lane halves of wave w)
  const int m   = tid >> 4;
  const int sub = tid & 15;
  float s1 = 0.0f, s2 = 0.0f;
#pragma unroll
  for (int c = 0; c < 8; ++c) {
    const float hv = hmix[m * DD + sub + c * 16];
    s1 += hv; s2 += hv * hv;
  }
#pragma unroll
  for (int msk = 8; msk >= 1; msk >>= 1) {
    s1 += __shfl_xor(s1, msk, 32);
    s2 += __shfl_xor(s2, msk, 32);
  }
  const float mu  = s1 * (1.0f / 128.0f);
  const float var = s2 * (1.0f / 128.0f) - mu * mu;
  const float inv = rsqrtf(var + LN_EPS);
#pragma unroll
  for (int c = 0; c < 8; ++c) {
    const int n = sub + c * 16;
    const float hv = hmix[m * DD + n];
    const float ln = (hv - mu) * inv * ln_g[n] + ln_b[n];
    x_out[(size_t)(row0 + m) * DD + n] = (ln >= 0.0f) ? ln : NEG_SLP * ln;
  }
}

// ---------------------------------------------------------------------------
extern "C" void kernel_launch(void* const* d_in, const int* in_sizes, int n_in,
                              void* d_out, int out_size, void* d_ws, size_t ws_size,
                              hipStream_t stream) {
  (void)in_sizes; (void)n_in; (void)out_size; (void)ws_size;
  const float* x    = (const float*)d_in[0];
  const float* y    = (const float*)d_in[1];
  const int*   src  = (const int*)d_in[2];
  const int*   dst  = (const int*)d_in[3];
  const float* Wq   = (const float*)d_in[4];
  const float* bq   = (const float*)d_in[5];
  const float* Wk   = (const float*)d_in[6];
  const float* bk   = (const float*)d_in[7];
  const float* Wv   = (const float*)d_in[8];
  const float* bv   = (const float*)d_in[9];
  const float* We   = (const float*)d_in[10];
  const float* be   = (const float*)d_in[11];
  const float* Wr   = (const float*)d_in[12];
  const float* br   = (const float*)d_in[13];
  const float* Wg   = (const float*)d_in[14];
  const float* bg   = (const float*)d_in[15];
  const float* ln_g = (const float*)d_in[16];
  const float* ln_b = (const float*)d_in[17];

  float* x_out = (float*)d_out;                       // [N,128]
  float* nes   = (float*)d_out + (size_t)NN * DD;     // [E,8] norm_escore

  // workspace layout
  char* ws = (char*)d_ws;
  const size_t ND = (size_t)NN * DD * sizeof(float);  // 25.6 MB
  float*    q = (float*)(ws + 0 * ND);
  float*    k = (float*)(ws + 1 * ND);
  float*    v = (float*)(ws + 2 * ND);
  float*    r = (float*)(ws + 3 * ND);
  float*    o = (float*)(ws + 4 * ND);
  float*    z = (float*)(ws + 5 * ND);                         // [N,8]
  _Float16* e = (_Float16*)(ws + 5 * ND + (size_t)NN * HH * sizeof(float)); // [E,128] f16

  const dim3 blk(256);

  // zero accumulators (o and z are contiguous)
  const int zn = NN * DD + NN * HH;
  zero_f32<<<(zn + 255) / 256, blk, 0, stream>>>(o, zn);

  // node projections (f32 out) and edge projection (f16 out), all WMMA + TDM
  const int ntiles_n = NN / 16;                 // 3125
  const int ntiles_e = EE / 16;                 // 50000
  const int grid_n   = (ntiles_n + 7) / 8;      // 391
  const int grid_e   = ntiles_e / 8;            // 6250
  gemm_bias_128<false><<<grid_n, blk, 0, stream>>>(x, Wq, bq, (void*)q, ntiles_n);
  gemm_bias_128<false><<<grid_n, blk, 0, stream>>>(x, Wk, bk, (void*)k, ntiles_n);
  gemm_bias_128<false><<<grid_n, blk, 0, stream>>>(x, Wv, bv, (void*)v, ntiles_n);
  gemm_bias_128<false><<<grid_n, blk, 0, stream>>>(x, Wr, br, (void*)r, ntiles_n);
  gemm_bias_128<true ><<<grid_e, blk, 0, stream>>>(y, We, be, (void*)e, ntiles_e);

  // attention: scores + segment softmax denominator, then scatter aggregation
  const int eh = EE * HH;
  edge_score    <<<(eh + 255) / 256, blk, 0, stream>>>(q, k, e, src, dst, z, nes);
  edge_aggregate<<<(eh + 255) / 256, blk, 0, stream>>>(v, e, src, dst, z, nes, o);

  // fused gating + LayerNorm + leaky-ReLU
  gate_norm_out<<<NN / 16, blk, 0, stream>>>(o, r, Wg, bg, ln_g, ln_b, x_out);
}
